// MultiHeadAttention_10411000726015
// MI455X (gfx1250) — matrix operand order repaired and checked by reading, not compiled, not run on hardware
//
#include <hip/hip_runtime.h>
#include <hip/hip_bf16.h>

// Problem constants (from reference)
#define BB   2
#define TT   2048
#define DIN  1024
#define DOUT 1024
#define NH   16
#define HDIM 64
#define BT   (BB * TT)                 // 4096 token rows
#define NQ   (BB * NH * TT * HDIM)     // elems per Q/K/V/ctx buffer (4,194,304)
#define NX   (BT * DIN)                // bf16 x copy (4,194,304)
#define NW   (DOUT * DIN)              // bf16 weight copy (1,048,576)

typedef __attribute__((ext_vector_type(16))) __bf16 v16bf;
typedef __attribute__((ext_vector_type(8)))  float  v8f;
typedef __attribute__((ext_vector_type(4)))  int    v4i;

union BF16x16 { unsigned u[8]; v16bf v; };

// ---- CDNA5 async global->LDS path (guarded; falls back to uint4 copies) ----
// Probe-verified: __builtin_amdgcn_global_load_async_to_lds_b128 takes
// (v4i* src, v4i* lds_dst, imm offset, imm cpol) with generic pointers.
#if defined(__gfx1250__) && __has_builtin(__builtin_amdgcn_global_load_async_to_lds_b128) && __has_builtin(__builtin_amdgcn_s_wait_asynccnt)
#define HAVE_ASYNC 1
#define GPTR(p) ((v4i*)(p))
#define LPTR(p) ((v4i*)(p))
#else
#define HAVE_ASYNC 0
#endif

__device__ __forceinline__ void async_wait() {
#if HAVE_ASYNC
  __builtin_amdgcn_s_wait_asynccnt(0);
#endif
}

// Cheap f32->bf16 (round-half-up; 2 VALU)
__device__ __forceinline__ unsigned short f2bf(float f) {
  return (unsigned short)((__float_as_uint(f) + 0x8000u) >> 16);
}

// Two f32 -> packed bf16 dword: round-half-up + one v_perm_b32 (3 VALU total)
__device__ __forceinline__ unsigned pack2bf(float x, float y) {
  unsigned ux = __float_as_uint(x) + 0x8000u;
  unsigned uy = __float_as_uint(y) + 0x8000u;
  // D = {uy[31:16], ux[31:16]} : sel bytes {7,6 | 3,2} from concat {s0=uy : s1=ux}
  return __builtin_amdgcn_perm(uy, ux, 0x07060302u);
}

// A-matrix 16x32 bf16 fragment:
// lanes 0-15: M=lane, K 0..15; lanes 16-31: M=lane-16, K 16..31
__device__ __forceinline__ v16bf load_A_frag(const unsigned short* base, int ld, int lane) {
  BF16x16 r;
  const unsigned short* rowp = base + (lane & 15) * ld;
  int kb = (lane & 16) ? 16 : 0;
#pragma unroll
  for (int i = 0; i < 8; ++i) {
    int k = kb + 2 * i;
    r.u[i] = *(const unsigned*)(rowp + k);
  }
  return r.v;
}

// B-matrix 32x16 bf16 fragment; LDS holds B transposed: base[n*ld + k] == B[k][n]
__device__ __forceinline__ v16bf load_B_frag(const unsigned short* base, int ld, int lane) {
  BF16x16 r;
  const unsigned short* rowp = base + (lane & 15) * ld;
  int kb = (lane & 16) ? 16 : 0;
#pragma unroll
  for (int i = 0; i < 8; ++i)
    r.u[i] = *(const unsigned*)(rowp + kb + 2 * i);
  return r.v;
}

// B fragment from row-major LDS tile: base[k*ld + n] == B[k][n] (V tiles)
__device__ __forceinline__ v16bf load_B_frag_rowmajor(const unsigned short* base, int ld, int lane) {
  BF16x16 r;
  int n  = lane & 15;
  int kb = (lane & 16) ? 16 : 0;
#pragma unroll
  for (int i = 0; i < 8; ++i) {
    unsigned lo = base[(kb + 2 * i)     * ld + n];
    unsigned hi = base[(kb + 2 * i + 1) * ld + n];
    r.u[i] = lo | (hi << 16);
  }
  return r.v;
}

__device__ __forceinline__ v8f wmma_bf16(v16bf a, v16bf b, v8f c) {
  return __builtin_amdgcn_wmma_f32_16x16x32_bf16(false, a, false, b, (short)0, c, false, false);
}

#define LDA 40   // 32 bf16 + 8 pad: 80B row stride (16B-aligned)

// Register-blocked compute step: wave owns 64(M) x 32(N) of the 128x128 tile.
// 4 A-frags + 2 B-frags -> 8 WMMAs per 32-deep K step.
__device__ __forceinline__ void gemm_step(const unsigned short* As, const unsigned short* Bs,
                                          int mw, int nw, int lane, v8f (&acc)[4][2]) {
  v16bf a[4], b[2];
#pragma unroll
  for (int ms = 0; ms < 4; ++ms)
    a[ms] = load_A_frag(As + (mw * 64 + ms * 16) * LDA, LDA, lane);
#pragma unroll
  for (int ns = 0; ns < 2; ++ns)
    b[ns] = load_B_frag(Bs + (nw * 32 + ns * 16) * LDA, LDA, lane);
#pragma unroll
  for (int ms = 0; ms < 4; ++ms)
#pragma unroll
    for (int ns = 0; ns < 2; ++ns)
      acc[ms][ns] = wmma_bf16(a[ms], b[ns], acc[ms][ns]);
}

// Stage a 128x32 bf16 tile into padded LDS (2 x b128 per thread, 256 threads)
__device__ __forceinline__ void stage_tile128x32(unsigned short* dstLds, const unsigned short* src,
                                                 size_t ldsrc, int tid) {
#if HAVE_ASYNC
#pragma unroll
  for (int j = 0; j < 2; ++j) {
    int idx = tid + j * 256, row = idx >> 2, c8 = (idx & 3) * 8;
    __builtin_amdgcn_global_load_async_to_lds_b128(
        GPTR(src + (size_t)row * ldsrc + c8), LPTR(dstLds + row * LDA + c8), 0, 0);
  }
#else
#pragma unroll
  for (int j = 0; j < 2; ++j) {
    int idx = tid + j * 256, row = idx >> 2, c8 = (idx & 3) * 8;
    *(uint4*)&dstLds[row * LDA + c8] = *(const uint4*)(src + (size_t)row * ldsrc + c8);
  }
#endif
}

// ---------------------------------------------------------------------------
// Kernel 0: one-pass fp32 -> bf16 conversion (x and the 4 weight matrices).
// Hoists ALL conversion VALU out of the GEMM hot loops; GEMM staging becomes
// pure GLOBAL_LOAD_ASYNC_TO_LDS_B128.
// ---------------------------------------------------------------------------
__global__ void __launch_bounds__(256) cvt_bf16_kernel(const float* __restrict__ src,
                                                       unsigned short* __restrict__ dst,
                                                       int n4) {
  int i = blockIdx.x * 256 + threadIdx.x;
  if (i < n4) {
    float4 v = ((const float4*)src)[i];
    uint2 o;
    o.x = pack2bf(v.x, v.y);
    o.y = pack2bf(v.z, v.w);
    ((uint2*)dst)[i] = o;
  }
}

// ---------------------------------------------------------------------------
// Kernel 1: Q/K/V projection.  out[bt,e] = sum_d x[bt,d] * W[e,d]
// 256 threads (8 waves), 128x128 tile, double-buffered LDS; A and B tiles are
// bf16 and streamed with async-to-LDS. Output bf16 in (B,H,T,HD).
// ---------------------------------------------------------------------------
__global__ void __launch_bounds__(256) qkv_gemm_kernel(
    const unsigned short* __restrict__ xb, const unsigned short* __restrict__ wqb,
    const unsigned short* __restrict__ wkb, const unsigned short* __restrict__ wvb,
    unsigned short* __restrict__ qb, unsigned short* __restrict__ kb,
    unsigned short* __restrict__ vb) {
  __shared__ unsigned short As[2][128 * LDA];
  __shared__ unsigned short Bs[2][128 * LDA];

  const int tid = threadIdx.x, lane = tid & 31, w = tid >> 5;
  const int mw = w >> 2, nw = w & 3;               // 2 x 4 wave grid
  const int bt0 = blockIdx.x * 128, e0 = blockIdx.y * 128;
  const unsigned short* wb = (blockIdx.z == 0) ? wqb : (blockIdx.z == 1) ? wkb : wvb;
  unsigned short* dst = (blockIdx.z == 0) ? qb : (blockIdx.z == 1) ? kb : vb;

  v8f acc[4][2] = {};

  stage_tile128x32(As[0], xb + (size_t)bt0 * DIN, DIN, tid);
  stage_tile128x32(Bs[0], wb + (size_t)e0  * DIN, DIN, tid);
  async_wait();
  __syncthreads();

  const int NSTEP = DIN / 32;
  for (int it = 0; it < NSTEP; ++it) {
    int cur = it & 1;
    if (it + 1 < NSTEP) {                          // stream next K-slab while computing
      stage_tile128x32(As[cur ^ 1], xb + (size_t)bt0 * DIN + (it + 1) * 32, DIN, tid);
      stage_tile128x32(Bs[cur ^ 1], wb + (size_t)e0  * DIN + (it + 1) * 32, DIN, tid);
    }
    gemm_step(As[cur], Bs[cur], mw, nw, lane, acc);
    async_wait();
    __syncthreads();
  }

  const int n = lane & 15, mh = (lane >> 4) & 1;
#pragma unroll
  for (int ms = 0; ms < 4; ++ms)
#pragma unroll
    for (int ns = 0; ns < 2; ++ns) {
      int e = e0 + nw * 32 + ns * 16 + n;
      int h = e >> 6, hd = e & 63;
#pragma unroll
      for (int r = 0; r < 8; ++r) {
        int btg = bt0 + mw * 64 + ms * 16 + mh * 8 + r;
        int b = btg >> 11, t = btg & (TT - 1);
        dst[(((size_t)(b * NH + h)) * TT + t) * HDIM + hd] = f2bf(acc[ms][ns][r]);
      }
    }
}

// ---------------------------------------------------------------------------
// Kernel 2: causal flash attention. Block = 4 waves = 64 query rows of (b,h).
// K/V chunks double-buffered via async-to-LDS; chunk k+1 streams during WMMA.
// Mask -inf first, then softmax(S / sqrt(HD)) — matches reference order.
// ---------------------------------------------------------------------------
#define LDQ 72   // 64 bf16 + 8 pad: 144B row stride (16B-aligned)

__device__ __forceinline__ void stage_tile64(unsigned short* dst, const unsigned short* src, int tid) {
#if HAVE_ASYNC
  for (int i = tid; i < 512; i += 128) {
    int r = i >> 3, q8 = i & 7;
    __builtin_amdgcn_global_load_async_to_lds_b128(
        GPTR(src + r * HDIM + q8 * 8), LPTR(dst + r * LDQ + q8 * 8), 0, 0);
  }
#else
  for (int i = tid; i < 512; i += 128) {
    int r = i >> 3, q8 = i & 7;
    *(uint4*)&dst[r * LDQ + q8 * 8] = *(const uint4*)(src + r * HDIM + q8 * 8);
  }
#endif
}

__global__ void __launch_bounds__(128) attn_kernel(
    const unsigned short* __restrict__ qb, const unsigned short* __restrict__ kb,
    const unsigned short* __restrict__ vb, unsigned short* __restrict__ ctx) {
  __shared__ unsigned short Qs[64 * LDQ];
  __shared__ unsigned short Ks[2][64 * LDQ];
  __shared__ unsigned short Vs[2][64 * LDQ];
  __shared__ unsigned short Ps[4 * 16 * LDQ];     // per-wave P transpose buffer

  const int tid = threadIdx.x, lane = tid & 31, w = tid >> 5;
  const int qc = blockIdx.x, h = blockIdx.y, b = blockIdx.z;
  const size_t head_base = ((size_t)(b * NH + h)) * TT * HDIM;

  stage_tile64(Qs, qb + head_base + (size_t)qc * 64 * HDIM, tid);
  async_wait();
  __syncthreads();
  const v16bf qf0 = load_A_frag(Qs + w * 16 * LDQ,      LDQ, lane);  // K = 0..31
  const v16bf qf1 = load_A_frag(Qs + w * 16 * LDQ + 32, LDQ, lane);  // K = 32..63

  stage_tile64(Ks[0], kb + head_base, tid);
  stage_tile64(Vs[0], vb + head_base, tid);
  async_wait();
  __syncthreads();

  v8f o[4] = {};
  float rm[8], rl[8];
#pragma unroll
  for (int r = 0; r < 8; ++r) { rm[r] = -__builtin_inff(); rl[r] = 0.f; }

  const float S2 = 0.125f * 1.44269504088896340736f;   // (1/sqrt(64)) * log2(e)
  const int n = lane & 15, mh = (lane >> 4) & 1;

  for (int kc = 0; kc <= qc; ++kc) {
    const int cur = kc & 1;
    if (kc < qc) {                                 // stream next chunk while computing
      stage_tile64(Ks[cur ^ 1], kb + head_base + (size_t)(kc + 1) * 64 * HDIM, tid);
      stage_tile64(Vs[cur ^ 1], vb + head_base + (size_t)(kc + 1) * 64 * HDIM, tid);
    }

    // S = Q K^T : wave computes 16 x 64 scores
    v8f s[4] = {};
#pragma unroll
    for (int nt = 0; nt < 4; ++nt) {
      v16bf k0f = load_B_frag(Ks[cur] + nt * 16 * LDQ,      LDQ, lane);
      v16bf k1f = load_B_frag(Ks[cur] + nt * 16 * LDQ + 32, LDQ, lane);
      s[nt] = wmma_bf16(qf0, k0f, s[nt]);
      s[nt] = wmma_bf16(qf1, k1f, s[nt]);
    }

    if (kc == qc) {                                // diagonal: causal mask (j > i)
      int ib = qc * 64 + w * 16 + mh * 8;
#pragma unroll
      for (int nt = 0; nt < 4; ++nt) {
        int j = kc * 64 + nt * 16 + n;
#pragma unroll
        for (int r = 0; r < 8; ++r)
          if (j > ib + r) s[nt][r] = -__builtin_inff();
      }
    }

    // Online softmax: row stats replicated across each 16-lane half (wave32)
#pragma unroll
    for (int r = 0; r < 8; ++r) {
      float cm = fmaxf(fmaxf(s[0][r], s[1][r]), fmaxf(s[2][r], s[3][r]));
#pragma unroll
      for (int msk = 8; msk >= 1; msk >>= 1)
        cm = fmaxf(cm, __shfl_xor(cm, msk, 16));
      float nm = fmaxf(rm[r], cm);
      float al = exp2f((rm[r] - nm) * S2);         // rm==-inf on first chunk -> 0
      float cs = 0.f;
#pragma unroll
      for (int nt = 0; nt < 4; ++nt) {
        float p = exp2f((s[nt][r] - nm) * S2);
        cs += p;
        Ps[(w * 16 + mh * 8 + r) * LDQ + nt * 16 + n] = f2bf(p);
      }
#pragma unroll
      for (int msk = 8; msk >= 1; msk >>= 1)
        cs += __shfl_xor(cs, msk, 16);
      rl[r] = rl[r] * al + cs;
      rm[r] = nm;
#pragma unroll
      for (int nt = 0; nt < 4; ++nt) o[nt][r] *= al;
    }
    __syncthreads();                               // P transpose buffer visible

    // O += P @ V
    v16bf pf0 = load_A_frag(Ps + w * 16 * LDQ,      LDQ, lane);
    v16bf pf1 = load_A_frag(Ps + w * 16 * LDQ + 32, LDQ, lane);
#pragma unroll
    for (int nt = 0; nt < 4; ++nt) {
      v16bf v0f = load_B_frag_rowmajor(Vs[cur] + nt * 16,            LDQ, lane);
      v16bf v1f = load_B_frag_rowmajor(Vs[cur] + 32 * LDQ + nt * 16, LDQ, lane);
      o[nt] = wmma_bf16(pf0, v0f, o[nt]);
      o[nt] = wmma_bf16(pf1, v1f, o[nt]);
    }
    async_wait();                                  // next chunk landed in LDS
    __syncthreads();
  }

  // Normalize and store ctx (bf16, (B,H,T,HD) = reference's raw reshape order)
  unsigned short* gc = ctx + head_base + (size_t)qc * 64 * HDIM;
#pragma unroll
  for (int r = 0; r < 8; ++r) {
    float inv = 1.0f / rl[r];
    int m = w * 16 + mh * 8 + r;
#pragma unroll
    for (int nt = 0; nt < 4; ++nt)
      gc[(size_t)m * HDIM + nt * 16 + n] = f2bf(o[nt][r] * inv);
  }
}

// ---------------------------------------------------------------------------
// Kernel 3: output projection. out[bt,e] = sum_d ctx_flat[bt,d]*Wo[e,d] + bo[e]
// ctx (B,H,T,HD) read FLAT as (B*T,1024) == reference's un-transposed view.
// Both tiles bf16 -> pure async-to-LDS staging.
// ---------------------------------------------------------------------------
__global__ void __launch_bounds__(256) out_gemm_kernel(
    const unsigned short* __restrict__ ctx, const unsigned short* __restrict__ wob,
    const float* __restrict__ bo, float* __restrict__ out) {
  __shared__ unsigned short As[2][128 * LDA];
  __shared__ unsigned short Bs[2][128 * LDA];

  const int tid = threadIdx.x, lane = tid & 31, w = tid >> 5;
  const int mw = w >> 2, nw = w & 3;
  const int bt0 = blockIdx.x * 128, e0 = blockIdx.y * 128;

  v8f acc[4][2] = {};

  stage_tile128x32(As[0], ctx + (size_t)bt0 * DOUT, DOUT, tid);
  stage_tile128x32(Bs[0], wob + (size_t)e0  * DOUT, DOUT, tid);
  async_wait();
  __syncthreads();

  const int NSTEP = DOUT / 32;
  for (int it = 0; it < NSTEP; ++it) {
    int cur = it & 1;
    if (it + 1 < NSTEP) {
      stage_tile128x32(As[cur ^ 1], ctx + (size_t)bt0 * DOUT + (it + 1) * 32, DOUT, tid);
      stage_tile128x32(Bs[cur ^ 1], wob + (size_t)e0  * DOUT + (it + 1) * 32, DOUT, tid);
    }
    gemm_step(As[cur], Bs[cur], mw, nw, lane, acc);
    async_wait();
    __syncthreads();
  }

  const int n = lane & 15, mh = (lane >> 4) & 1;
#pragma unroll
  for (int ms = 0; ms < 4; ++ms)
#pragma unroll
    for (int ns = 0; ns < 2; ++ns) {
      int e = e0 + nw * 32 + ns * 16 + n;
      float bias = bo[e];
#pragma unroll
      for (int r = 0; r < 8; ++r) {
        int btg = bt0 + mw * 64 + ms * 16 + mh * 8 + r;
        out[(size_t)btg * DOUT + e] = acc[ms][ns][r] + bias;
      }
    }
}

// ---------------------------------------------------------------------------
extern "C" void kernel_launch(void* const* d_in, const int* in_sizes, int n_in,
                              void* d_out, int out_size, void* d_ws, size_t ws_size,
                              hipStream_t stream) {
  const float* x  = (const float*)d_in[0];
  const float* Wq = (const float*)d_in[1];
  const float* Wk = (const float*)d_in[2];
  const float* Wv = (const float*)d_in[3];
  const float* Wo = (const float*)d_in[4];
  const float* bo = (const float*)d_in[5];
  float* out = (float*)d_out;

  // Workspace (bf16): Q | K | V | ctx | x | Wq | Wk | Wv | Wo  (48 MB total)
  unsigned short* qbuf = (unsigned short*)d_ws;
  unsigned short* kbuf = qbuf + NQ;
  unsigned short* vbuf = kbuf + NQ;
  unsigned short* cbuf = vbuf + NQ;
  unsigned short* xb   = cbuf + NQ;
  unsigned short* wqb  = xb + NX;
  unsigned short* wkb  = wqb + NW;
  unsigned short* wvb  = wkb + NW;
  unsigned short* wob  = wvb + NW;

  // One-time fp32 -> bf16 conversion pass (bandwidth-bound, hits L2)
  cvt_bf16_kernel<<<(NX / 4 + 255) / 256, 256, 0, stream>>>(x,  xb,  NX / 4);
  cvt_bf16_kernel<<<(NW / 4 + 255) / 256, 256, 0, stream>>>(Wq, wqb, NW / 4);
  cvt_bf16_kernel<<<(NW / 4 + 255) / 256, 256, 0, stream>>>(Wk, wkb, NW / 4);
  cvt_bf16_kernel<<<(NW / 4 + 255) / 256, 256, 0, stream>>>(Wv, wvb, NW / 4);
  cvt_bf16_kernel<<<(NW / 4 + 255) / 256, 256, 0, stream>>>(Wo, wob, NW / 4);

  qkv_gemm_kernel<<<dim3(BT / 128, DOUT / 128, 3), 256, 0, stream>>>(
      xb, wqb, wkb, wvb, qbuf, kbuf, vbuf);
  attn_kernel<<<dim3(TT / 64, NH, BB), 128, 0, stream>>>(qbuf, kbuf, vbuf, cbuf);
  out_gemm_kernel<<<dim3(BT / 128, DOUT / 128), 256, 0, stream>>>(cbuf, wob, bo, out);
}
